// ComplexHoloLinear_12008728559944
// MI455X (gfx1250) — matrix operand order repaired and checked by reading, not compiled, not run on hardware
//
#include <hip/hip_runtime.h>
#include <hip/hip_bf16.h>
#include <math.h>

// Problem constants (match reference)
#define NNZ    4194304
#define IN_F   4096
#define OUT_F  4096
#define BATCH  4
#define SEQ    2048
#define NROWS  (BATCH * SEQ)        // 8192 flattened tokens
#define WELEM  (OUT_F * IN_F)       // 16777216 dense weight elements

typedef __bf16 bf16_t;
typedef __attribute__((ext_vector_type(4)))  bf16_t bf16x4;
typedef __attribute__((ext_vector_type(8)))  bf16_t bf16x8;
typedef __attribute__((ext_vector_type(16))) bf16_t bf16x16;
typedef __attribute__((ext_vector_type(4)))  float  f32x4;
typedef __attribute__((ext_vector_type(8)))  float  f32x8;
typedef int v4i __attribute__((vector_size(16)));   // builtin's b128 payload type

#define AS1 __attribute__((address_space(1)))
#define AS3 __attribute__((address_space(3)))

// CDNA5 async global->LDS path, guarded so the kernel compiles on any toolchain.
#if defined(__HIP_DEVICE_COMPILE__) && defined(__gfx1250__) && \
    __has_builtin(__builtin_amdgcn_global_load_async_to_lds_b128) && \
    __has_builtin(__builtin_amdgcn_s_wait_asynccnt)
#define USE_ASYNC 1
#else
#define USE_ASYNC 0
#endif

__device__ __forceinline__ void cp_b128_to_lds(void* lds_dst, const void* g_src) {
#if USE_ASYNC
  __builtin_amdgcn_global_load_async_to_lds_b128(
      (AS1 v4i*)(__UINTPTR_TYPE__)g_src,
      (AS3 v4i*)lds_dst, /*offset=*/0, /*cpol=*/0);
#else
  *(bf16x8*)lds_dst = *(const bf16x8*)g_src;
#endif
}

__device__ __forceinline__ void wait_lds_fill() {
#if USE_ASYNC
  __builtin_amdgcn_s_wait_asynccnt(0);
#endif
}

// ---------------- Phase 1: zero fp32 W workspace ----------------
__global__ __launch_bounds__(256) void zero_f4(float4* p, int n4) {
  int i = blockIdx.x * 256 + threadIdx.x;
  if (i < n4) p[i] = make_float4(0.f, 0.f, 0.f, 0.f);
}

// ---------------- Phase 2: COO scatter (duplicates sum) --------
__global__ __launch_bounds__(256) void scatter_coo(
    const int* __restrict__ rows, const int* __restrict__ cols,
    const float* __restrict__ wr, const float* __restrict__ wi,
    float* __restrict__ Wr32, float* __restrict__ Wi32) {
  int e = blockIdx.x * 256 + threadIdx.x;
  if (e < NNZ) {
    size_t off = (size_t)rows[e] * IN_F + (size_t)cols[e];
    atomicAdd(Wr32 + off, wr[e]);
    atomicAdd(Wi32 + off, wi[e]);
  }
}

// ---------------- Phase 3: fp32 -> bf16 conversion -------------
__global__ __launch_bounds__(256) void cvt_f32_bf16(
    const float* __restrict__ src, bf16_t* __restrict__ dst, int n) {
  int i = (blockIdx.x * 256 + threadIdx.x) * 4;
  if (i < n) {
    f32x4 v = *(const f32x4*)(src + i);
    bf16x4 o;
#pragma unroll
    for (int j = 0; j < 4; ++j) o[j] = (bf16_t)v[j];
    *(bf16x4*)(dst + i) = o;
  }
}

// ---------------- Phase 4: fused dual GEMM + phase epilogue ----
// out[n,r] = sum_k X[n,k]*Wr[r,k] + cos(phase[n/SEQ]) * sum_k X[n,k]*Wi[r,k]
// Block tile: 128(M) x 64(N). 256 threads = 8 waves in a 4(M) x 2(N) grid,
// each wave owns a 32x32 tile = 2x2 WMMA 16x16 tiles, two accumulators each.
#define BM 128
#define BN 64
#define BK 32
#define LDA 40   // padded row stride (bf16 elems): 80B, keeps 16B alignment, kills bank conflicts
#define LDB 40
#define KT (IN_F / BK)  // 128 k-steps

__global__ __launch_bounds__(256) void gemm_dual_bf16(
    const bf16_t* __restrict__ X, const bf16_t* __restrict__ Wr,
    const bf16_t* __restrict__ Wi, const float* __restrict__ phase,
    float* __restrict__ out) {
  __shared__ __align__(16) bf16_t sA[2][BM * LDA];
  __shared__ __align__(16) bf16_t sBr[2][BN * LDB];
  __shared__ __align__(16) bf16_t sBi[2][BN * LDB];

  const int tid     = threadIdx.x;
  const int lane    = tid & 31;
  const int wave    = tid >> 5;
  const int wm      = wave >> 1;          // 0..3 : M position of wave
  const int wn      = wave & 1;           // 0..1 : N position of wave
  const int half16  = lane & 15;
  const bool hiHalf = lane >= 16;

  const int rowBase = blockIdx.x * BM;    // token rows
  const int colBase = blockIdx.y * BN;    // output features

  // Stage one BK-slice of A (128x32), Br (64x32), Bi (64x32) into LDS buffer `buf`.
  auto load_tiles = [&](int buf, int kt) {
    const int k0 = kt * BK;
#pragma unroll
    for (int i = 0; i < 2; ++i) {         // A: 512 x b128 chunks, 2 per thread
      int idx = tid + i * 256;
      int row = idx >> 2, ch = idx & 3;   // 4 x 16B chunks per 32-elem row
      cp_b128_to_lds(&sA[buf][row * LDA + ch * 8],
                     X + (size_t)(rowBase + row) * IN_F + k0 + ch * 8);
    }
    {
      int row = tid >> 2, ch = tid & 3;   // B: 256 chunks each, 1 per thread
      cp_b128_to_lds(&sBr[buf][row * LDB + ch * 8],
                     Wr + (size_t)(colBase + row) * IN_F + k0 + ch * 8);
      cp_b128_to_lds(&sBi[buf][row * LDB + ch * 8],
                     Wi + (size_t)(colBase + row) * IN_F + k0 + ch * 8);
    }
  };

  f32x8 accR[2][2] = {};
  f32x8 accI[2][2] = {};

  load_tiles(0, 0);
  for (int kt = 0; kt < KT; ++kt) {
    wait_lds_fill();        // this wave's async fills for buffer kt&1 done
    __syncthreads();        // all waves' fills visible in LDS
    if (kt + 1 < KT) load_tiles((kt + 1) & 1, kt + 1);  // prefetch next buffer
    const int buf = kt & 1;

    // A fragment, this kernel's k order (B uses the same): lanes 0-15 hold K 0..15 of row M=lane,
    // lanes 16-31 hold K 16..31 (contiguous). Two ds_read_b128 per tile.
    bf16x16 a[2], br[2], bi[2];
#pragma unroll
    for (int tm = 0; tm < 2; ++tm) {
      const bf16_t* p = &sA[buf][(wm * 32 + tm * 16 + half16) * LDA + (hiHalf ? 16 : 0)];
      bf16x8 lo = *(const bf16x8*)p;
      bf16x8 hi = *(const bf16x8*)(p + 8);
      a[tm] = __builtin_shufflevector(lo, hi, 0,1,2,3,4,5,6,7,8,9,10,11,12,13,14,15);
    }
    // B fragment, same k order as A: lane holds column N=lane%16, K 0..15 (lanes 0-15)
    // or K 16..31 (lanes 16-31), contiguous -> two ds_read_b128 per tile.
#pragma unroll
    for (int tn = 0; tn < 2; ++tn) {
      const bf16_t* pr = &sBr[buf][(wn * 32 + tn * 16 + half16) * LDB + (hiHalf ? 16 : 0)];
      bf16x8 rlo = *(const bf16x8*)pr;
      bf16x8 rhi = *(const bf16x8*)(pr + 8);
      br[tn] = __builtin_shufflevector(rlo, rhi, 0,1,2,3,4,5,6,7,8,9,10,11,12,13,14,15);
      const bf16_t* pi = &sBi[buf][(wn * 32 + tn * 16 + half16) * LDB + (hiHalf ? 16 : 0)];
      bf16x8 ilo = *(const bf16x8*)pi;
      bf16x8 ihi = *(const bf16x8*)(pi + 8);
      bi[tn] = __builtin_shufflevector(ilo, ihi, 0,1,2,3,4,5,6,7,8,9,10,11,12,13,14,15);
    }

#pragma unroll
    for (int tm = 0; tm < 2; ++tm)
#pragma unroll
      for (int tn = 0; tn < 2; ++tn) {
        accR[tm][tn] = __builtin_amdgcn_wmma_f32_16x16x32_bf16(
            false, a[tm], false, br[tn], (short)0, accR[tm][tn], false, false);
        accI[tm][tn] = __builtin_amdgcn_wmma_f32_16x16x32_bf16(
            false, a[tm], false, bi[tn], (short)0, accI[tm][tn], false, false);
      }
  }

  // Epilogue: cos(phase) is uniform across the whole 128-row block (2048 % 128 == 0).
  const float c = cosf(phase[rowBase / SEQ]);
  // C/D layout: VGPR v, lanes 0-15 -> (M=v, N=lane); lanes 16-31 -> (M=v+8, N=lane-16).
#pragma unroll
  for (int tm = 0; tm < 2; ++tm)
#pragma unroll
    for (int tn = 0; tn < 2; ++tn) {
      const int col = colBase + wn * 32 + tn * 16 + half16;
#pragma unroll
      for (int v = 0; v < 8; ++v) {
        const int row = rowBase + wm * 32 + tm * 16 + v + (hiHalf ? 8 : 0);
        out[(size_t)row * OUT_F + col] = accR[tm][tn][v] + c * accI[tm][tn][v];
      }
    }
}

// ---------------- Launch ----------------
extern "C" void kernel_launch(void* const* d_in, const int* in_sizes, int n_in,
                              void* d_out, int out_size, void* d_ws, size_t ws_size,
                              hipStream_t stream) {
  const float* x       = (const float*)d_in[0];
  const int*   rows    = (const int*)d_in[1];
  const int*   cols    = (const int*)d_in[2];
  const float* w_real  = (const float*)d_in[3];
  const float* w_imag  = (const float*)d_in[4];
  const float* phase   = (const float*)d_in[5];
  float*       out     = (float*)d_out;

  // Workspace layout (256 MB):
  //   Wr32[16M f32] | Wi32[16M f32] | Xb[32M bf16] | Wrb[16M bf16] | Wib[16M bf16]
  float*  Wr32 = (float*)d_ws;
  float*  Wi32 = Wr32 + (size_t)WELEM;
  bf16_t* Xb   = (bf16_t*)(Wi32 + (size_t)WELEM);
  bf16_t* Wrb  = Xb + (size_t)NROWS * IN_F;
  bf16_t* Wib  = Wrb + (size_t)WELEM;

  // 1) zero the fp32 accumulation buffers (2*16M floats = 8M float4)
  const int n4 = (2 * WELEM) / 4;
  zero_f4<<<(n4 + 255) / 256, 256, 0, stream>>>((float4*)Wr32, n4);

  // 2) scatter COO -> dense fp32 (atomics, mostly L2-resident: 128MB < 192MB L2)
  scatter_coo<<<(NNZ + 255) / 256, 256, 0, stream>>>(rows, cols, w_real, w_imag,
                                                     Wr32, Wi32);

  // 3) convert x and W to bf16 for the matrix cores
  const int nx = NROWS * IN_F;
  cvt_f32_bf16<<<(nx / 4 + 255) / 256, 256, 0, stream>>>(x, Xb, nx);
  cvt_f32_bf16<<<(WELEM / 4 + 255) / 256, 256, 0, stream>>>(Wr32, Wrb, WELEM);
  cvt_f32_bf16<<<(WELEM / 4 + 255) / 256, 256, 0, stream>>>(Wi32, Wib, WELEM);

  // 4) fused dual GEMM with phase epilogue
  dim3 grid(NROWS / BM, OUT_F / BN);   // 64 x 64 blocks
  gemm_dual_bf16<<<grid, 256, 0, stream>>>(Xb, Wrb, Wib, phase, out);
}
